// EpisodicEchoModelV2_75892072121015
// MI455X (gfx1250) — matrix operand order repaired and checked by reading, not compiled, not run on hardware
//
#include <hip/hip_runtime.h>
#include <hip/hip_bf16.h>
#include <math.h>

typedef __attribute__((ext_vector_type(16))) _Float16 v16h;
typedef __attribute__((ext_vector_type(8)))  _Float16 v8h;
typedef __attribute__((ext_vector_type(8)))  float    v8f;

#define BB 2
#define SS 1024
#define DD 128
#define D2 256
#define NN 256
#define LL 2
#define HH 2
#define VV 32000
#define PHI_F 1.61803398874989485f

// ---------------- WMMA helpers (CDNA5 16x16x32 f16 -> f32) ----------------

__device__ __forceinline__ v8f wmma_f16(v16h a, v16h b, v8f c) {
  return __builtin_amdgcn_wmma_f32_16x16x32_f16(false, a, false, b, (short)0, c,
                                                false, false);
}

// A-matrix fragment (16x32, f16). lane l: row = l&15; halves 0..7 -> K=base..base+7,
// halves 8..15 -> K=base+16..base+23, base = (l&16)?8:0.
__device__ __forceinline__ v16h load_a_frag(const _Float16* base, int ld, int k0, int lane) {
  const _Float16* p = base + (size_t)(lane & 15) * ld + k0 + ((lane & 16) ? 8 : 0);
  v8h lo = *(const v8h*)p;
  v8h hi = *(const v8h*)(p + 16);
  v16h a;
#pragma unroll
  for (int i = 0; i < 8; ++i) { a[i] = lo[i]; a[i + 8] = hi[i]; }
  return a;
}

// B-matrix fragment (32x16, f16), operand stored N-major (row n, contiguous K).
// lane l: col = l&15; halves 0..7 -> K=base..base+7,
// halves 8..15 -> K=base+16..base+23, base = (l&16)?8:0.
__device__ __forceinline__ v16h load_b_frag(const _Float16* base, int ld, int k0, int lane) {
  const _Float16* p = base + (size_t)(lane & 15) * ld + k0 + ((lane & 16) ? 8 : 0);
  v8h lo = *(const v8h*)p;
  v8h hi = *(const v8h*)(p + 16);
  v16h b;
#pragma unroll
  for (int i = 0; i < 8; ++i) { b[i] = lo[i]; b[i + 8] = hi[i]; }
  return b;
}

// ---------------- K1: embedding gather + euler scan ----------------
// nr = cos(th_r+th_i), ni = sin(th_r+th_i); sum = (hr+hi)/wl + 2b + 2*t*phi.
__global__ __launch_bounds__(128) void euler_scan_k(const int* __restrict__ ids,
                                                    const float* __restrict__ embed,
                                                    float* __restrict__ sr,
                                                    float* __restrict__ si,
                                                    float* __restrict__ x,
                                                    _Float16* __restrict__ Kh,
                                                    _Float16* __restrict__ KhT) {
  const int b = blockIdx.x;
  const int d = threadIdx.x;
  float hr = 0.0f, hi = 0.0f;
  for (int t = 0; t < SS; ++t) {
    const int id = ids[b * SS + t];
    const float w  = embed[(size_t)id * D2 + d];
    const float be = embed[(size_t)id * D2 + DD + d];
    const float wl = 1.0f + fabsf(w);
    const float sum = (hr + hi) / wl + 2.0f * be + 2.0f * (float)t * PHI_F;
    float sv, cv;
    sincosf(sum, &sv, &cv);
    hr = cv; hi = sv;
    const size_t idx = ((size_t)b * SS + t) * DD + d;
    sr[idx] = cv; si[idx] = sv; x[idx] = cv + sv;
    const size_t kidx = ((size_t)b * SS + t) * D2;
    Kh[kidx + d]      = (_Float16)cv;
    Kh[kidx + DD + d] = (_Float16)sv;
    KhT[((size_t)b * D2 + d) * SS + t]      = (_Float16)cv;
    KhT[((size_t)b * D2 + DD + d) * SS + t] = (_Float16)sv;
  }
}

// ---------------- K2: per-layer Q = concat(cos(th_r), cos(th_i)) as f16 ------
__global__ __launch_bounds__(256) void compute_q_k(const float* __restrict__ sr,
                                                   const float* __restrict__ si,
                                                   const float* __restrict__ wq,
                                                   const float* __restrict__ bq,
                                                   _Float16* __restrict__ Qh) {
  const int tid = blockIdx.x * 256 + threadIdx.x;
  const int d = tid & 127;
  const int s = (tid >> 7) & 1023;
  const int b = (tid >> 17) & 1;
  const int h = tid >> 18;
  const float wl = 1.0f + fabsf(wq[h * DD + d]);
  const float bv = bq[h * DD + d];
  const float tphi = (float)s * PHI_F;
  const size_t sidx = ((size_t)b * SS + s) * DD + d;
  const float thr = sr[sidx] / wl + bv + tphi;
  const float thi = si[sidx] / wl + bv + tphi;
  const size_t qidx = ((size_t)(h * BB + b) * SS + s) * D2;
  Qh[qidx + d]      = (_Float16)cosf(thr);
  Qh[qidx + DD + d] = (_Float16)cosf(thi);
}

// ---------------- K3: flash-attention, 1 wave per 16-query tile --------------
__global__ __launch_bounds__(32) void attn_k(const _Float16* __restrict__ Qh,
                                             const _Float16* __restrict__ Kh,
                                             const _Float16* __restrict__ KhT,
                                             float* __restrict__ retr) {
  __shared__ _Float16 pLDS[16 * 32];
  const int lane = threadIdx.x & 31;
  const int qi = blockIdx.x & 63;
  const int bh = blockIdx.x >> 6;
  const int h = bh & 1;
  const int b = bh >> 1;

  const _Float16* Qbase  = Qh  + ((size_t)(h * BB + b) * SS + qi * 16) * D2;
  const _Float16* Kbase  = Kh  + (size_t)b * SS * D2;
  const _Float16* KTbase = KhT + (size_t)b * D2 * SS;

  v16h qfrag[8];
#pragma unroll
  for (int kk = 0; kk < 8; ++kk) qfrag[kk] = load_a_frag(Qbase, D2, kk * 32, lane);

  v8f Oacc[16];
#pragma unroll
  for (int t = 0; t < 16; ++t)
#pragma unroll
    for (int v = 0; v < 8; ++v) Oacc[t][v] = 0.0f;

  float m[8], lsum[8];
#pragma unroll
  for (int v = 0; v < 8; ++v) { m[v] = -3.0e38f; lsum[v] = 0.0f; }

  const int mbase = (lane & 16) ? 8 : 0;
  const int nchunks = (qi >> 1) + 1;
  const float scale = 0.0625f;  // 1/sqrt(2D)=1/16

  for (int c = 0; c < nchunks; ++c) {
    const int jt0 = 2 * c, jt1 = 2 * c + 1;
    const bool t1v = (jt1 <= qi);
    v8f s0, s1;
#pragma unroll
    for (int v = 0; v < 8; ++v) { s0[v] = 0.0f; s1[v] = 0.0f; }
    {
      const _Float16* kb = Kbase + (size_t)jt0 * 16 * D2;
#pragma unroll
      for (int kk = 0; kk < 8; ++kk)
        s0 = wmma_f16(qfrag[kk], load_b_frag(kb, D2, kk * 32, lane), s0);
    }
    if (t1v) {
      const _Float16* kb = Kbase + (size_t)jt1 * 16 * D2;
#pragma unroll
      for (int kk = 0; kk < 8; ++kk)
        s1 = wmma_f16(qfrag[kk], load_b_frag(kb, D2, kk * 32, lane), s1);
    }
    const int key0 = jt0 * 16 + (lane & 15);
    const int key1 = jt1 * 16 + (lane & 15);
    float corr[8];
#pragma unroll
    for (int v = 0; v < 8; ++v) {
      const int q = qi * 16 + v + mbase;
      float a0 = s0[v] * scale;
      if (jt0 == qi && key0 > q) a0 = -3.0e38f;
      if (q == 0 && jt0 == 0 && key0 == 0) a0 = 0.0f;   // scores[:,0,0]=0
      float a1 = t1v ? (s1[v] * scale) : -3.0e38f;
      if (t1v && jt1 == qi && key1 > q) a1 = -3.0e38f;
      float mx = fmaxf(a0, a1);
#pragma unroll
      for (int off = 8; off >= 1; off >>= 1) mx = fmaxf(mx, __shfl_xor(mx, off, 32));
      const float mn = fmaxf(m[v], mx);
      const float cr = __expf(m[v] - mn);
      const float p0 = __expf(a0 - mn);
      const float p1 = __expf(a1 - mn);
      float rs = p0 + p1;
#pragma unroll
      for (int off = 8; off >= 1; off >>= 1) rs += __shfl_xor(rs, off, 32);
      lsum[v] = lsum[v] * cr + rs;
      m[v] = mn;
      corr[v] = cr;
      pLDS[(v + mbase) * 32 + (lane & 15)]      = (_Float16)p0;
      pLDS[(v + mbase) * 32 + 16 + (lane & 15)] = (_Float16)p1;
    }
    __syncthreads();
    v16h pf;  // re-layout D->A through LDS
    {
      const _Float16* pp = &pLDS[(lane & 15) * 32 + ((lane & 16) ? 8 : 0)];
#pragma unroll
      for (int i = 0; i < 8; ++i) { pf[i] = pp[i]; pf[i + 8] = pp[i + 16]; }
    }
#pragma unroll
    for (int nt = 0; nt < 16; ++nt) {
#pragma unroll
      for (int v = 0; v < 8; ++v) Oacc[nt][v] *= corr[v];
      const _Float16* vb = KTbase + (size_t)nt * 16 * SS;
      Oacc[nt] = wmma_f16(pf, load_b_frag(vb, SS, c * 32, lane), Oacc[nt]);
    }
    __syncthreads();
  }

  float* rbase = retr + ((size_t)(h * BB + b) * SS + qi * 16) * D2;
#pragma unroll
  for (int nt = 0; nt < 16; ++nt)
#pragma unroll
    for (int v = 0; v < 8; ++v) {
      const int q = v + mbase;
      rbase[(size_t)q * D2 + nt * 16 + (lane & 15)] = Oacc[nt][v] / lsum[v];
    }
}

// ---------------- K4: EMA scan over S + head combine -------------------------
__global__ __launch_bounds__(256) void ema_combine_k(const float* __restrict__ retr,
                                                     const float* __restrict__ alpha_l,
                                                     float* __restrict__ comb) {
  const int tid = threadIdx.x;
  const int b = tid >> 7;
  const int d = tid & 127;
  const float a0 = 1.0f / (1.0f + __expf(-alpha_l[0]));
  const float a1 = 1.0f / (1.0f + __expf(-alpha_l[1]));
  const float* r0 = retr + (size_t)b * SS * D2;
  const float* r1 = retr + (size_t)(BB + b) * SS * D2;
  float e0l = 0.f, e0h = 0.f, e1l = 0.f, e1h = 0.f;
  for (int t = 0; t < SS; ++t) {
    const size_t o = (size_t)t * D2 + d;
    e0l += a0 * (r0[o] - e0l);
    e0h += a0 * (r0[o + DD] - e0h);
    e1l += a1 * (r1[o] - e1l);
    e1h += a1 * (r1[o + DD] - e1h);
    comb[((size_t)b * SS + t) * DD + d] = e0l + e0h + e1l + e1h;
  }
}

// ---------------- K5: resonant FFN sincos sums (one wave per (b,s,n)) --------
__global__ __launch_bounds__(256) void ffn_sums_k(const float* __restrict__ x,
                                                  const float* __restrict__ W,
                                                  const float* __restrict__ Bp,
                                                  const float* __restrict__ ac,
                                                  const float* __restrict__ as_,
                                                  float* __restrict__ csum,
                                                  float* __restrict__ ssum) {
  const int lane = threadIdx.x & 31;
  const int wid = blockIdx.x * 8 + (threadIdx.x >> 5);
  const int n = wid & (NN - 1);
  const int bs = wid >> 8;
  float cs = 0.0f, sn = 0.0f;
#pragma unroll
  for (int i = 0; i < 4; ++i) {
    const int d = lane + 32 * i;
    const size_t wi = (size_t)n * DD + d;
    const float th = x[(size_t)bs * DD + d] / (1.0f + fabsf(W[wi])) + Bp[wi];
    float s, cq;
    sincosf(th, &s, &cq);
    cs += cq * ac[wi];
    sn += s * as_[wi];
  }
#pragma unroll
  for (int off = 16; off >= 1; off >>= 1) {
    cs += __shfl_xor(cs, off, 32);
    sn += __shfl_xor(sn, off, 32);
  }
  if (lane == 0) {
    csum[(size_t)bs * NN + n] = cs;
    ssum[(size_t)bs * NN + n] = sn;
  }
}

// ---------------- K6: proj + silu + echo + residual update -------------------
__global__ __launch_bounds__(128) void layer_update_k(const float* __restrict__ comb,
                                                      const float* __restrict__ wout,
                                                      const float* __restrict__ bout,
                                                      const float* __restrict__ csum,
                                                      const float* __restrict__ ssum,
                                                      const float* __restrict__ pc,
                                                      const float* __restrict__ ps,
                                                      const float* __restrict__ esp,
                                                      const float* __restrict__ rsp,
                                                      float* __restrict__ x) {
  __shared__ float lc[NN], lsn[NN];
  const int bs = blockIdx.x;
  const int d = threadIdx.x;
  lc[d]        = csum[(size_t)bs * NN + d];
  lc[d + DD]   = csum[(size_t)bs * NN + d + DD];
  lsn[d]       = ssum[(size_t)bs * NN + d];
  lsn[d + DD]  = ssum[(size_t)bs * NN + d + DD];
  __syncthreads();
  float acc = 0.0f;
  const float* pcr = pc + (size_t)d * NN;
  const float* psr = ps + (size_t)d * NN;
#pragma unroll 4
  for (int n = 0; n < NN; ++n) acc += lc[n] * pcr[n] + lsn[n] * psr[n];
  const float res = acc / (1.0f + __expf(-acc));  // silu
  const float th = comb[(size_t)bs * DD + d] / (1.0f + fabsf(wout[d])) + bout[d];
  float sv, cv;
  sincosf(th, &sv, &cv);
  x[(size_t)bs * DD + d] += esp[0] * (cv + sv) + rsp[0] * res;
}

// ---------------- K7: f32 -> f16 conversion ----------------------------------
__global__ void conv_f16_k(const float* __restrict__ src, _Float16* __restrict__ dst, int n) {
  const int i = blockIdx.x * blockDim.x + threadIdx.x;
  if (i < n) dst[i] = (_Float16)src[i];
}

// ---------------- K8: logits GEMM (2048 x 32000, K=128), WMMA f16 ------------
// One wave computes a 16x64 tile (A frags reused across 4 N-tiles).
__global__ __launch_bounds__(256) void out_gemm_k(const _Float16* __restrict__ xh,
                                                  const _Float16* __restrict__ wh,
                                                  float* __restrict__ out) {
  const int lane = threadIdx.x & 31;
  const int wid = blockIdx.x * 8 + (threadIdx.x >> 5);  // 64000 waves
  const int mt = wid / 500;   // 128 M-tiles
  const int ns = wid % 500;   // 500 N-strips of 64
  const _Float16* ab = xh + (size_t)mt * 16 * DD;
  v16h af[4];
#pragma unroll
  for (int k = 0; k < 4; ++k) af[k] = load_a_frag(ab, DD, k * 32, lane);
  const int mbase = (lane & 16) ? 8 : 0;
#pragma unroll
  for (int t = 0; t < 4; ++t) {
    const int n0 = ns * 64 + t * 16;
    const _Float16* bb = wh + (size_t)n0 * DD;
    v8f acc;
#pragma unroll
    for (int v = 0; v < 8; ++v) acc[v] = 0.0f;
#pragma unroll
    for (int k = 0; k < 4; ++k)
      acc = wmma_f16(af[k], load_b_frag(bb, DD, k * 32, lane), acc);
#pragma unroll
    for (int v = 0; v < 8; ++v) {
      const int mrow = mt * 16 + v + mbase;
      out[(size_t)mrow * VV + n0 + (lane & 15)] = acc[v];
    }
  }
}

// ---------------- host orchestration -----------------------------------------
extern "C" void kernel_launch(void* const* d_in, const int* in_sizes, int n_in,
                              void* d_out, int out_size, void* d_ws, size_t ws_size,
                              hipStream_t stream) {
  (void)in_sizes; (void)n_in; (void)out_size; (void)ws_size;
  const int*   ids   = (const int*)  d_in[0];
  const float* embed = (const float*)d_in[1];
  const float* wq    = (const float*)d_in[2];
  const float* bq    = (const float*)d_in[3];
  const float* alpha = (const float*)d_in[4];
  const float* wout  = (const float*)d_in[5];
  const float* bout  = (const float*)d_in[6];
  const float* ffnW  = (const float*)d_in[7];
  const float* ffnB  = (const float*)d_in[8];
  const float* acs   = (const float*)d_in[9];
  const float* asn   = (const float*)d_in[10];
  const float* pcs   = (const float*)d_in[11];
  const float* psn   = (const float*)d_in[12];
  const float* esc   = (const float*)d_in[13];
  const float* rsc   = (const float*)d_in[14];
  const float* outp  = (const float*)d_in[15];

  char* ws = (char*)d_ws;
  const size_t MB1 = 1ull << 20;
  float*    sr   = (float*)(ws + 0 * MB1);   // 1 MB
  float*    si   = (float*)(ws + 1 * MB1);   // 1 MB
  float*    x    = (float*)(ws + 2 * MB1);   // 1 MB
  float*    comb = (float*)(ws + 3 * MB1);   // 1 MB
  float*    csum = (float*)(ws + 4 * MB1);   // 2 MB
  float*    ssum = (float*)(ws + 6 * MB1);   // 2 MB
  float*    retr = (float*)(ws + 8 * MB1);   // 4 MB (H*B*S*2D f32)
  _Float16* Kh   = (_Float16*)(ws + 12 * MB1); // 1 MB
  _Float16* KhT  = (_Float16*)(ws + 13 * MB1); // 1 MB
  _Float16* Qh   = (_Float16*)(ws + 14 * MB1); // 2 MB
  _Float16* xh   = (_Float16*)(ws + 16 * MB1); // 0.5 MB
  _Float16* wh   = (_Float16*)(ws + 17 * MB1); // 8 MB  (total ~25 MB)

  euler_scan_k<<<BB, DD, 0, stream>>>(ids, embed, sr, si, x, Kh, KhT);
  conv_f16_k<<<(VV * DD + 255) / 256, 256, 0, stream>>>(outp, wh, VV * DD);

  for (int l = 0; l < LL; ++l) {
    compute_q_k<<<(HH * BB * SS * DD) / 256, 256, 0, stream>>>(
        sr, si, wq + l * HH * DD, bq + l * HH * DD, Qh);
    attn_k<<<BB * HH * (SS / 16), 32, 0, stream>>>(Qh, Kh, KhT, retr);
    ema_combine_k<<<1, 256, 0, stream>>>(retr, alpha + l * HH, comb);
    ffn_sums_k<<<(BB * SS * NN) / 8, 256, 0, stream>>>(
        x, ffnW + (size_t)l * NN * DD, ffnB + (size_t)l * NN * DD,
        acs + (size_t)l * NN * DD, asn + (size_t)l * NN * DD, csum, ssum);
    layer_update_k<<<BB * SS, DD, 0, stream>>>(
        comb, wout + l * DD, bout + l * DD, csum, ssum,
        pcs + (size_t)l * DD * NN, psn + (size_t)l * DD * NN,
        esc + l, rsc + l, x);
  }

  conv_f16_k<<<(BB * SS * DD + 255) / 256, 256, 0, stream>>>(x, xh, BB * SS * DD);
  out_gemm_k<<<(128 * 500) / 8, 256, 0, stream>>>(xh, wh, (float*)d_out);
}
